// MultiHeadAttention_83863531421896
// MI455X (gfx1250) — matrix operand order repaired and checked by reading, not compiled, not run on hardware
//
#include <hip/hip_runtime.h>
#include <stdint.h>

#define D_MODEL 1024
#define SEQ     2048
#define NHEAD   16
#define HDIM    64
#define BATCH   2
#define MROWS   (BATCH*SEQ)   // 4096
#define N_QKV   (3*D_MODEL)   // 3072

typedef __attribute__((ext_vector_type(16))) __bf16 v16bf;
typedef __attribute__((ext_vector_type(8)))  float  v8f;

struct __align__(16) U4 { uint32_t x, y, z, w; };

union FragU { v16bf v; U4 q[2]; };

__device__ inline uint16_t f2bf(float f) {
  union { float f; uint32_t u; } a; a.f = f;
  uint32_t u = a.u + 0x7FFFu + ((a.u >> 16) & 1u);   // round-to-nearest-even
  return (uint16_t)(u >> 16);
}

__device__ inline v8f wmma_bf16(v16bf a, v16bf b, v8f c) {
  // (neg_a, A, neg_b, B, c_mod, C, reuse_a, reuse_b)
  return __builtin_amdgcn_wmma_f32_16x16x32_bf16(false, a, false, b, (short)0, c, false, false);
}

// A fragment: 16x32 (MxK) bf16 from row-major A (leading dim lda, elements).
// Lane L: m=L%16, half=L/16. elems 0..7 -> k=8*half+0..7 ; elems 8..15 -> k=16+8*half+0..7
__device__ inline v16bf load_a_frag(const uint16_t* A, int lda) {
  int lane = threadIdx.x & 31;
  int m = lane & 15, half = lane >> 4;
  const uint16_t* base = A + (size_t)m * lda + 8 * half;
  FragU f;
  f.q[0] = *(const U4*)(base);
  f.q[1] = *(const U4*)(base + 16);
  return f.v;
}

// B fragment: 32x16 (KxN) bf16 from N-major storage Bt[n*ldb + k].
// Lane L: n=L%16, half=L/16. elems 0..7 -> k=8*half+0..7 ; elems 8..15 -> k=16+8*half+0..7
__device__ inline v16bf load_b_frag(const uint16_t* Bt, int ldb) {
  int lane = threadIdx.x & 31;
  int n = lane & 15, half = lane >> 4;
  const uint16_t* base = Bt + (size_t)n * ldb + 8 * half;
  FragU f;
  f.q[0] = *(const U4*)(base);
  f.q[1] = *(const U4*)(base + 16);
  return f.v;
}

// ---------------- f32 -> bf16 conversion ----------------
__global__ __launch_bounds__(256) void cvt_f32_bf16(const float* __restrict__ in,
                                                    uint16_t* __restrict__ out, int n) {
  int i = blockIdx.x * 256 + threadIdx.x;
  if (i < n) out[i] = f2bf(in[i]);
}

// ---------------- QKV projection GEMM ----------------
// X: (4096,1024) bf16 row-major. W: (3072,1024) bf16 row-major (= B^T, N-major).
// 64x64 output per wave (4x4 wmma tiles): 32 FLOP/byte of fragment traffic.
// Output scattered: Q,K as (B*H, S, 64) bf16; V transposed as (B*H, 64, S) bf16.
__device__ inline void qkv_store_tile(v8f acc, int rm, int cn, const float* __restrict__ bias,
                                      uint16_t* __restrict__ Qb, uint16_t* __restrict__ Kb,
                                      uint16_t* __restrict__ Vt) {
  int lane = threadIdx.x & 31;
  int nl = lane & 15, g = lane >> 4;
  int h = cn / 192;          // head (3*hd = 192 columns per head)
  int c = cn % 192;          // position within head's qkv block (tile-aligned)
  float bv = bias[cn + nl];
#pragma unroll
  for (int r = 0; r < 8; ++r) {
    int sg = rm + r + 8 * g;                 // global token row 0..4095
    int b = sg >> 11, s = sg & (SEQ - 1);
    uint16_t o = f2bf(acc[r] + bv);
    if (c < 64) {
      Qb[((size_t)(b * NHEAD + h) * SEQ + s) * HDIM + c + nl] = o;
    } else if (c < 128) {
      Kb[((size_t)(b * NHEAD + h) * SEQ + s) * HDIM + (c - 64) + nl] = o;
    } else {
      Vt[((size_t)(b * NHEAD + h) * HDIM + (c - 128) + nl) * SEQ + s] = o;
    }
  }
}

__global__ __launch_bounds__(128) void qkv_gemm(const uint16_t* __restrict__ X,
                                                const uint16_t* __restrict__ W,
                                                const float* __restrict__ bias,
                                                uint16_t* __restrict__ Qb,
                                                uint16_t* __restrict__ Kb,
                                                uint16_t* __restrict__ Vt) {
  int wid = threadIdx.x >> 5;
  int gid = blockIdx.x * 4 + wid;            // 3072 waves
  int tM = gid / (N_QKV / 64);               // 64 row supertiles
  int tN = gid % (N_QKV / 64);               // 48 col supertiles
  int row0 = tM * 64, col0 = tN * 64;
  v8f acc[4][4] = {};
  for (int kk = 0; kk < D_MODEL; kk += 32) {
    v16bf a[4], b[4];
#pragma unroll
    for (int i = 0; i < 4; ++i)
      a[i] = load_a_frag(X + (size_t)(row0 + 16 * i) * D_MODEL + kk, D_MODEL);
#pragma unroll
    for (int j = 0; j < 4; ++j)
      b[j] = load_b_frag(W + (size_t)(col0 + 16 * j) * D_MODEL + kk, D_MODEL);
#pragma unroll
    for (int i = 0; i < 4; ++i)
#pragma unroll
      for (int j = 0; j < 4; ++j)
        acc[i][j] = wmma_bf16(a[i], b[j], acc[i][j]);
  }
#pragma unroll
  for (int i = 0; i < 4; ++i)
#pragma unroll
    for (int j = 0; j < 4; ++j)
      qkv_store_tile(acc[i][j], row0 + 16 * i, col0 + 16 * j, bias, Qb, Kb, Vt);
}

// ---------------- Flash attention ----------------
// One wave per (batch*head, 32-query tile): K/V fragments amortized over two
// query tiles (16 wmma per 32-key step vs 8 fragment loads).
// Q,K: (BH,S,64) bf16; Vt: (BH,64,S) bf16. AO: (B*S, 1024) bf16 row-major.
__global__ __launch_bounds__(128) void attn_flash(const uint16_t* __restrict__ Q,
                                                  const uint16_t* __restrict__ K,
                                                  const uint16_t* __restrict__ Vt,
                                                  uint16_t* __restrict__ AO) {
  __shared__ __align__(16) uint16_t Plds[4][2][16 * 32];  // 8KB per block
  int wid = threadIdx.x >> 5;
  int lane = threadIdx.x & 31;
  int gid = blockIdx.x * 4 + wid;            // 2048 waves
  int bh = gid >> 6;                         // 0..31
  int qt = gid & 63;
  int q0 = qt * 32;
  const uint16_t* Qh = Q + (size_t)bh * SEQ * HDIM;
  const uint16_t* Kh = K + (size_t)bh * SEQ * HDIM;
  const uint16_t* Vh = Vt + (size_t)bh * HDIM * SEQ;

  v16bf qa[2][2];
#pragma unroll
  for (int t = 0; t < 2; ++t) {
    qa[t][0] = load_a_frag(Qh + (size_t)(q0 + 16 * t) * HDIM, HDIM);
    qa[t][1] = load_a_frag(Qh + (size_t)(q0 + 16 * t) * HDIM + 32, HDIM);
  }

  v8f acc[2][4] = {};
  float mrow[2][8], lrow[2][8];
#pragma unroll
  for (int t = 0; t < 2; ++t)
#pragma unroll
    for (int r = 0; r < 8; ++r) { mrow[t][r] = -3.0e38f; lrow[t][r] = 0.0f; }
  const float sc = 0.125f;                   // 1/sqrt(64)
  int g = lane >> 4, nl = lane & 15;

  for (int k0 = 0; k0 < SEQ; k0 += 32) {
    v16bf kb[2][2];
#pragma unroll
    for (int u = 0; u < 2; ++u) {
      kb[u][0] = load_b_frag(Kh + (size_t)(k0 + 16 * u) * HDIM, HDIM);
      kb[u][1] = load_b_frag(Kh + (size_t)(k0 + 16 * u) * HDIM + 32, HDIM);
    }
#pragma unroll
    for (int t = 0; t < 2; ++t) {
      v8f z = {};
      v8f s0 = wmma_bf16(qa[t][0], kb[0][0], z); s0 = wmma_bf16(qa[t][1], kb[0][1], s0);
      v8f s1 = wmma_bf16(qa[t][0], kb[1][0], z); s1 = wmma_bf16(qa[t][1], kb[1][1], s1);

#pragma unroll
      for (int r = 0; r < 8; ++r) {
        float a = s0[r] * sc, b = s1[r] * sc;
        float mx = fmaxf(a, b);
        mx = fmaxf(mx, __shfl_xor(mx, 1));
        mx = fmaxf(mx, __shfl_xor(mx, 2));
        mx = fmaxf(mx, __shfl_xor(mx, 4));
        mx = fmaxf(mx, __shfl_xor(mx, 8));   // row max across 16-lane group
        float mn = fmaxf(mrow[t][r], mx);
        float alpha = __expf(mrow[t][r] - mn);
        float e0 = __expf(a - mn), e1 = __expf(b - mn);
        float rs = e0 + e1;
        rs += __shfl_xor(rs, 1);
        rs += __shfl_xor(rs, 2);
        rs += __shfl_xor(rs, 4);
        rs += __shfl_xor(rs, 8);             // row sum
        lrow[t][r] = lrow[t][r] * alpha + rs;
        mrow[t][r] = mn;
#pragma unroll
        for (int j = 0; j < 4; ++j) acc[t][j][r] *= alpha;
        // C-layout -> row-major 16x32 P tile in LDS.
        int row = r + 8 * g;
        Plds[wid][t][row * 32 + nl]      = f2bf(e0);
        Plds[wid][t][row * 32 + 16 + nl] = f2bf(e1);
      }
    }
    v16bf pa0 = load_a_frag(&Plds[wid][0][0], 32);
    v16bf pa1 = load_a_frag(&Plds[wid][1][0], 32);
#pragma unroll
    for (int j = 0; j < 4; ++j) {
      v16bf vb = load_b_frag(Vh + (size_t)(16 * j) * SEQ + k0, SEQ);
      acc[0][j] = wmma_bf16(pa0, vb, acc[0][j]);
      acc[1][j] = wmma_bf16(pa1, vb, acc[1][j]);
    }
  }

  int b = bh >> 4, h = bh & 15;
#pragma unroll
  for (int t = 0; t < 2; ++t)
#pragma unroll
    for (int r = 0; r < 8; ++r) {
      int s = q0 + 16 * t + r + 8 * g;
      float inv = 1.0f / lrow[t][r];
      size_t rowbase = ((size_t)b * SEQ + s) * D_MODEL + h * HDIM + nl;
      AO[rowbase + 0]  = f2bf(acc[t][0][r] * inv);
      AO[rowbase + 16] = f2bf(acc[t][1][r] * inv);
      AO[rowbase + 32] = f2bf(acc[t][2][r] * inv);
      AO[rowbase + 48] = f2bf(acc[t][3][r] * inv);
    }
}

// ---------------- Output projection GEMM ----------------
// 64x64 output per wave (4x4 wmma tiles), f32 epilogue + bias into d_out.
__global__ __launch_bounds__(128) void out_gemm(const uint16_t* __restrict__ A,
                                                const uint16_t* __restrict__ W,
                                                const float* __restrict__ bias,
                                                float* __restrict__ Out) {
  int wid = threadIdx.x >> 5;
  int lane = threadIdx.x & 31;
  int gid = blockIdx.x * 4 + wid;            // 1024 waves
  int tM = gid >> 4;                         // 64 row supertiles
  int tN = gid & 15;                         // 16 col supertiles
  int row0 = tM * 64, col0 = tN * 64;
  v8f acc[4][4] = {};
  for (int kk = 0; kk < D_MODEL; kk += 32) {
    v16bf a[4], b[4];
#pragma unroll
    for (int i = 0; i < 4; ++i)
      a[i] = load_a_frag(A + (size_t)(row0 + 16 * i) * D_MODEL + kk, D_MODEL);
#pragma unroll
    for (int j = 0; j < 4; ++j)
      b[j] = load_b_frag(W + (size_t)(col0 + 16 * j) * D_MODEL + kk, D_MODEL);
#pragma unroll
    for (int i = 0; i < 4; ++i)
#pragma unroll
      for (int j = 0; j < 4; ++j)
        acc[i][j] = wmma_bf16(a[i], b[j], acc[i][j]);
  }
  int nl = lane & 15, g = lane >> 4;
#pragma unroll
  for (int i = 0; i < 4; ++i)
#pragma unroll
    for (int j = 0; j < 4; ++j) {
      int rm = row0 + 16 * i;
      int cn = col0 + 16 * j;
      float bv = bias[cn + nl];
#pragma unroll
      for (int r = 0; r < 8; ++r) {
        int row = rm + r + 8 * g;
        Out[(size_t)row * D_MODEL + cn + nl] = acc[i][j][r] + bv;
      }
    }
}

extern "C" void kernel_launch(void* const* d_in, const int* in_sizes, int n_in,
                              void* d_out, int out_size, void* d_ws, size_t ws_size,
                              hipStream_t stream) {
  const float* x     = (const float*)d_in[0];
  const float* w_qkv = (const float*)d_in[1];
  const float* b_qkv = (const float*)d_in[2];
  const float* w_out = (const float*)d_in[3];
  const float* b_out = (const float*)d_in[4];
  float* out = (float*)d_out;

  uint8_t* ws = (uint8_t*)d_ws;
  size_t off = 0;
  uint16_t* xb  = (uint16_t*)(ws + off); off += (size_t)MROWS * D_MODEL * 2;   // 8 MB
  uint16_t* wqb = (uint16_t*)(ws + off); off += (size_t)N_QKV * D_MODEL * 2;   // 6 MB
  uint16_t* wob = (uint16_t*)(ws + off); off += (size_t)D_MODEL * D_MODEL * 2; // 2 MB
  uint16_t* Qb  = (uint16_t*)(ws + off); off += (size_t)BATCH * NHEAD * SEQ * HDIM * 2; // 8 MB
  uint16_t* Kb  = (uint16_t*)(ws + off); off += (size_t)BATCH * NHEAD * SEQ * HDIM * 2; // 8 MB
  uint16_t* Vt  = (uint16_t*)(ws + off); off += (size_t)BATCH * NHEAD * HDIM * SEQ * 2; // 8 MB
  uint16_t* AO  = (uint16_t*)(ws + off); off += (size_t)MROWS * D_MODEL * 2;   // 8 MB

  int nx = MROWS * D_MODEL;       // 4194304
  int nwq = N_QKV * D_MODEL;      // 3145728
  int nwo = D_MODEL * D_MODEL;    // 1048576
  cvt_f32_bf16<<<(nx  + 255) / 256, 256, 0, stream>>>(x,     xb,  nx);
  cvt_f32_bf16<<<(nwq + 255) / 256, 256, 0, stream>>>(w_qkv, wqb, nwq);
  cvt_f32_bf16<<<(nwo + 255) / 256, 256, 0, stream>>>(w_out, wob, nwo);

  // 3072 waves = (4096/64)*(3072/64); 4 waves per 128-thread block
  qkv_gemm<<<768, 128, 0, stream>>>(xb, wqb, b_qkv, Qb, Kb, Vt);

  // 2048 waves = 32 heads * 64 query tiles of 32
  attn_flash<<<512, 128, 0, stream>>>(Qb, Kb, Vt, AO);

  // 1024 waves = (4096/64)*(1024/64)
  out_gemm<<<256, 128, 0, stream>>>(AO, wob, b_out, out);
}
